// Model_70222715289880
// MI455X (gfx1250) — compile-verified, matrix operand order checked by reading, not run on hardware
//
#include <hip/hip_runtime.h>
#include <hip/hip_bf16.h>
#include <math.h>

typedef __attribute__((ext_vector_type(16))) _Float16 v16h;
typedef __attribute__((ext_vector_type(8)))  _Float16 v8h;
typedef __attribute__((ext_vector_type(8)))  float    v8f;

#define B_    8
#define K_    64
#define TOPK_ 10
#define N_R_  4
#define L_R_  128
#define H_    512
#define D_    128
#define D2_   640

// ---- output layout (element offsets into d_out, all float) ----
#define O_SCORES   0L
#define O_S_R      80L                       // + B*TOPK
#define O_H_R      163920L                   // + B*TOPK*N_R*H
#define O_R_A      21135440L                 // + B*TOPK*N_R*L_R*H
#define O_R_O      21176400L
#define O_MASK     21217360L
#define O_REVBATCH 21217680L
#define O_REVMASK  21227920L
#define O_RATINGS  21238160L
#define O_REVEXT   21238240L
#define O_IDX      21248480L

// -------- Stage 0: stage Q (padded to 16 rows) and W^T in f16 ------------
__global__ void prep_q(const float* __restrict__ q, _Float16* __restrict__ qh) {
  const int i = blockIdx.x * blockDim.x + threadIdx.x;   // 16*D2
  if (i >= 16 * D2_) return;
  const int m = i / D2_;
  qh[i] = (m < B_) ? (_Float16)q[i] : (_Float16)0.0f;
}

__global__ void prep_wT(const float* __restrict__ W, _Float16* __restrict__ whT) {
  const int i = blockIdx.x * blockDim.x + threadIdx.x;   // D2*D2, i = e*D2 + d
  if (i >= D2_ * D2_) return;
  const int e = i / D2_;
  const int d = i - e * D2_;
  whT[i] = (_Float16)W[(size_t)d * D2_ + e];             // whT[e][d] = W[d][e]
}

// -------- Stage 1: T(16x640) = pad16(Q) @ W via v_wmma_f32_16x16x32_f16 ----
// One wave per 16x16 tile of T. 16-bit A layout: lanes 0-15 row M=lane,
// K runs [hi*8..hi*8+7] and [16+hi*8..+7] are contiguous -> two b128 loads.
// B mirrored from whT (N-major), also two b128 loads. Inner loop:
// 4x global_load_b128 + 1 wmma, branch-free.
__global__ __launch_bounds__(32)
void qw_wmma(const _Float16* __restrict__ qh,    // (16, D2) f16 padded
             const _Float16* __restrict__ whT,   // (D2, D2) f16, N-major
             float* __restrict__ t_out) {        // (16, D2) f32 padded
  const int lane = threadIdx.x;
  const int n0   = blockIdx.x * 16;
  const int mn   = lane & 15;
  const int hi   = lane >> 4;
  const _Float16* qrow = qh  + (size_t)mn * D2_ + hi * 8;
  const _Float16* wrow = whT + (size_t)(n0 + mn) * D2_ + hi * 8;
  v8f acc = {};
#pragma unroll 4
  for (int kt = 0; kt < D2_ / 32; ++kt) {
    const int k0 = kt * 32;
    const v8h alo = *(const v8h*)(qrow + k0);
    const v8h ahi = *(const v8h*)(qrow + k0 + 16);
    const v8h blo = *(const v8h*)(wrow + k0);
    const v8h bhi = *(const v8h*)(wrow + k0 + 16);
    v16h a, b;
#pragma unroll
    for (int j = 0; j < 8; ++j) {
      a[j] = alo[j]; a[j + 8] = ahi[j];
      b[j] = blo[j]; b[j + 8] = bhi[j];
    }
    acc = __builtin_amdgcn_wmma_f32_16x16x32_f16(false, a, false, b,
                                                 (short)0, acc, false, false);
  }
  // C/D layout: VGPR r -> row hi*8+r, column = lane&15
#pragma unroll
  for (int r = 0; r < 8; ++r)
    t_out[(size_t)(hi * 8 + r) * D2_ + n0 + mn] = acc[r];
}

// -------- Stage 2: scores[b,k] = tanh(dot(T[b,:], a_s_r[b,k,:])) ----------
__global__ void scores_kernel(const float* __restrict__ t,      // (16, D2)
                              const float* __restrict__ a_s_r,  // (B, K, D2)
                              float* __restrict__ scores) {     // (B*K)
  const int gid  = blockIdx.x * blockDim.x + threadIdx.x;
  const int wave = gid >> 5;
  const int lane = gid & 31;
  if (wave >= B_ * K_) return;
  const int b = wave / K_;
  const float4* tr = (const float4*)(t + (size_t)b * D2_);
  const float4* rr = (const float4*)(a_s_r + (size_t)wave * D2_);
  float s = 0.f;
#pragma unroll
  for (int e = lane; e < D2_ / 4; e += 32) {   // 5 iterations of b128 loads
    const float4 x = tr[e], y = rr[e];
    s += x.x * y.x + x.y * y.y + x.z * y.z + x.w * y.w;
  }
#pragma unroll
  for (int off = 16; off > 0; off >>= 1) s += __shfl_down(s, off, 32);
  if (lane == 0) scores[wave] = tanhf(s);
}

// -------- Stage 3: per-row top-10 of 64 (stable, lowest-index on ties) ----
__global__ void topk_kernel(const float* __restrict__ scores,  // (B*K)
                            float* __restrict__ out_scores,
                            float* __restrict__ out_idx_f,
                            int* __restrict__ idx_ws) {
  const int b = threadIdx.x;
  if (b >= B_) return;
  float s[K_];
  for (int k = 0; k < K_; ++k) s[k] = scores[b * K_ + k];
  for (int t = 0; t < TOPK_; ++t) {
    int best = 0; float bv = s[0];
    for (int k = 1; k < K_; ++k)
      if (s[k] > bv) { bv = s[k]; best = k; }   // strict > : first max wins
    out_scores[b * TOPK_ + t] = bv;
    out_idx_f [b * TOPK_ + t] = (float)best;
    idx_ws    [b * TOPK_ + t] = best;
    s[best] = -INFINITY;
  }
}

// -------- Stage 4: gathers (the bandwidth-bound part) ---------------------
// src (B,K,inner) f32 -> dst (B,TOPK,inner), 16B vectorized. inner4 = 1<<sh.
__global__ void gather_f4(const float4* __restrict__ src,
                          float4* __restrict__ dst,
                          const int* __restrict__ idx,
                          int sh, long total4) {
  long i = (long)blockIdx.x * blockDim.x + threadIdx.x;
  const long stride = (long)gridDim.x * blockDim.x;
  for (; i < total4; i += stride) {
    const long bt = i >> sh;
    const long r  = i & (((long)1 << sh) - 1);
    const int  b  = (int)(bt / TOPK_);
    const int  k  = idx[bt];
    dst[i] = src[(((long)(b * K_ + k)) << sh) + r];
  }
}

__global__ void gather_i2f(const int* __restrict__ src,
                           float* __restrict__ dst,
                           const int* __restrict__ idx,
                           int sh, int total) {
  const int i = blockIdx.x * blockDim.x + threadIdx.x;
  if (i >= total) return;
  const int bt = i >> sh;
  const int r  = i & ((1 << sh) - 1);
  const int b  = bt / TOPK_;
  const int k  = idx[bt];
  dst[i] = (float)src[((b * K_ + k) << sh) + r];
}

extern "C" void kernel_launch(void* const* d_in, const int* in_sizes, int n_in,
                              void* d_out, int out_size, void* d_ws, size_t ws_size,
                              hipStream_t stream) {
  const float* a_s_q        = (const float*)d_in[0];
  const float* a_s_r        = (const float*)d_in[1];
  const float* W            = (const float*)d_in[2];
  const float* s_r          = (const float*)d_in[3];
  const float* h_r          = (const float*)d_in[4];
  const float* r_a          = (const float*)d_in[5];
  const float* r_o          = (const float*)d_in[6];
  const float* rev_asp_mask = (const float*)d_in[7];
  const int*   rev_batch    = (const int*)d_in[8];
  const float* rev_mask     = (const float*)d_in[9];
  const int*   ratings      = (const int*)d_in[10];
  const int*   rev_extend   = (const int*)d_in[11];
  float* out = (float*)d_out;

  // workspace layout (16B-aligned regions)
  char* ws = (char*)d_ws;
  _Float16* qh   = (_Float16*)ws;                           // 16*640 f16  = 20480 B
  _Float16* whT  = (_Float16*)(ws + 20480);                 // 640*640 f16 = 819200 B
  float*    t_ws = (float*)(ws + 20480 + 819200);           // 16*640 f32  = 40960 B
  float*    scores_ws = (float*)(ws + 20480 + 819200 + 40960);   // 512 f32
  int*      idx_ws    = (int*)(ws + 20480 + 819200 + 40960 + 2048);  // 80 i32

  // Stage 0: f16 staging (branch-free WMMA operand layout)
  prep_q <<<(16 * D2_ + 255) / 256, 256, 0, stream>>>(a_s_q, qh);
  prep_wT<<<(D2_ * D2_ + 255) / 256, 256, 0, stream>>>(W, whT);

  // Stage 1: 40 tiles x 1 wave, WMMA f16->f32
  qw_wmma<<<D2_ / 16, 32, 0, stream>>>(qh, whT, t_ws);

  // Stage 2: 512 waves
  scores_kernel<<<(B_ * K_ * 32 + 255) / 256, 256, 0, stream>>>(t_ws, a_s_r, scores_ws);

  // Stage 3
  topk_kernel<<<1, 32, 0, stream>>>(scores_ws, out + O_SCORES, out + O_IDX, idx_ws);

  // Stage 4: vectorized gathers (inner sizes are all powers of two)
  auto g4 = [&](const float* src, long ooff, int inner, long count_bt) {
    const int  inner4 = inner / 4;
    const int  sh     = __builtin_ctz(inner4);
    const long total4 = count_bt * inner4;
    int blocks = (int)((total4 + 255) / 256);
    if (blocks > 65535) blocks = 65535;           // grid-stride covers the rest
    gather_f4<<<blocks, 256, 0, stream>>>((const float4*)src,
                                          (float4*)(out + ooff),
                                          idx_ws, sh, total4);
  };
  const long BT = (long)B_ * TOPK_;
  g4(s_r,          O_S_R,     N_R_ * H_,        BT);   // inner 2048
  g4(h_r,          O_H_R,     N_R_ * L_R_ * H_, BT);   // inner 262144 (dominant)
  g4(r_a,          O_R_A,     N_R_ * D_,        BT);   // inner 512
  g4(r_o,          O_R_O,     N_R_ * D_,        BT);   // inner 512
  g4(rev_asp_mask, O_MASK,    N_R_,             BT);   // inner 4
  g4(rev_mask,     O_REVMASK, L_R_,             BT);   // inner 128

  auto gi = [&](const int* src, long ooff, int inner) {
    const int sh    = __builtin_ctz(inner);
    const int total = (int)(BT * inner);
    gather_i2f<<<(total + 255) / 256, 256, 0, stream>>>(src, out + ooff,
                                                        idx_ws, sh, total);
  };
  gi(rev_batch,  O_REVBATCH, L_R_);
  gi(ratings,    O_RATINGS,  1);
  gi(rev_extend, O_REVEXT,   L_R_);
}
